// attention_net_76682346102902
// MI455X (gfx1250) — compile-verified, matrix operand order checked by reading, not run on hardware
//
#include <hip/hip_runtime.h>
#include <hip/hip_bf16.h>
#include <math.h>

typedef __attribute__((ext_vector_type(16))) _Float16 v16h;
typedef __attribute__((ext_vector_type(8)))  _Float16 v8h;
typedef __attribute__((ext_vector_type(8)))  float    v8f;

#define BATCH   8
#define A_N     1614
#define TOPN    6
#define PAD_PX  224
#define OUT_PX  224
#define IMG_PX  448

// ---------------------------------------------------------------------------
// Zero a run of f16 (used to clear padded NHWC activation buffers each call).
// ---------------------------------------------------------------------------
__global__ void zero_f16(_Float16* __restrict__ p, long n) {
  const long t = (long)blockIdx.x * blockDim.x + threadIdx.x;
  if (t < n) p[t] = (_Float16)0.f;
}

// ---------------------------------------------------------------------------
// NCHW f32 -> zero-padded NHWC f16: out shape (B, H+2, W+2, C), 1-px border.
// ---------------------------------------------------------------------------
template<int C, int H, int W>
__global__ void pack_act(const float* __restrict__ in, _Float16* __restrict__ out) {
  const long t = (long)blockIdx.x * blockDim.x + threadIdx.x;
  if (t >= (long)BATCH * C * H * W) return;
  const int b = (int)(t / (C * H * W));
  const int r = (int)(t % (C * H * W));
  const int c = r / (H * W);
  const int p = r % (H * W);
  const int y = p / W, x = p % W;
  out[(((long)b * (H + 2) + y + 1) * (W + 2) + (x + 1)) * C + c] = (_Float16)in[t];
}

// ---------------------------------------------------------------------------
// OIHW f32 -> [tap][oc][ic] f16  (tap = ky*3+kx; ic contiguous for b128 loads)
// ---------------------------------------------------------------------------
template<int Cin, int Cout>
__global__ void pack_wgt(const float* __restrict__ w, _Float16* __restrict__ out) {
  const long t = (long)blockIdx.x * blockDim.x + threadIdx.x;
  if (t >= (long)Cout * Cin * 9) return;
  const int oc  = (int)(t / (Cin * 9));
  const int r   = (int)(t % (Cin * 9));
  const int ic  = r / 9;
  const int tap = r % 9;
  out[((long)tap * Cout + oc) * Cin + ic] = (_Float16)w[t];
}

// ---------------------------------------------------------------------------
// Implicit-GEMM 3x3 conv + bias + ReLU on pre-packed f16 operands.
// A: padded NHWC f16 (B, HinP, WinP, Cin)   (pad already holds zeros)
// B: [tap][oc][ic] f16
// One wave per 16(pixel) x 16(out-channel) tile; f32 WMMA accumulation.
// Every lane's fragment half-group is a contiguous, 16B-aligned b128 load:
//   i<8  -> K = 8*khigh + i          (at base + k0)
//   i>=8 -> K = 16 + 8*khigh + (i-8) (at base + k0 + 16)
// Out-of-range M rows read clamped (valid) pixels and are never stored.
// All dims are template constants -> no runtime integer division.
// ---------------------------------------------------------------------------
template<int Cin, int HinP, int WinP, int Cout, int Hout, int Wout, int STRIDE>
__global__ __launch_bounds__(32)
void conv3x3_wmma(const _Float16* __restrict__ inP,
                  const _Float16* __restrict__ wP,
                  const float* __restrict__ bias,
                  float* __restrict__ outNCHW,
                  _Float16* __restrict__ outNHWC) {
  constexpr int M      = BATCH * Hout * Wout;
  constexpr int Mtiles = (M + 15) / 16;
  constexpr int hw     = Hout * Wout;

  const int lane   = threadIdx.x;
  const int tile_m = blockIdx.x % Mtiles;
  const int tile_n = blockIdx.x / Mtiles;
  const int khigh  = lane >> 4;
  const int msub   = lane & 15;
  const int oc     = tile_n * 16 + (lane & 15);

  int pm = tile_m * 16 + msub;
  if (pm >= M) pm = M - 1;                       // clamp: loads stay valid
  const int b  = pm / hw;
  const int rm = pm % hw;
  const int oy = rm / Wout;
  const int ox = rm % Wout;

  v8f acc = {0.f, 0.f, 0.f, 0.f, 0.f, 0.f, 0.f, 0.f};

  for (int tap = 0; tap < 9; ++tap) {
    const int ky = tap / 3, kx = tap % 3;
    const int py = oy * STRIDE + ky;             // padded coords, always in range
    const int px = ox * STRIDE + kx;
    const _Float16* abase = inP + (((long)b * HinP + py) * WinP + px) * Cin + khigh * 8;
    const _Float16* bbase = wP + ((long)tap * Cout + oc) * Cin + khigh * 8;
#pragma unroll 2
    for (int k0 = 0; k0 < Cin; k0 += 32) {
      const v8h alo = *(const v8h*)(abase + k0);
      const v8h ahi = *(const v8h*)(abase + k0 + 16);
      const v8h blo = *(const v8h*)(bbase + k0);
      const v8h bhi = *(const v8h*)(bbase + k0 + 16);
      const v16h afrag = __builtin_shufflevector(alo, ahi,
          0,1,2,3,4,5,6,7,8,9,10,11,12,13,14,15);
      const v16h bfrag = __builtin_shufflevector(blo, bhi,
          0,1,2,3,4,5,6,7,8,9,10,11,12,13,14,15);
      acc = __builtin_amdgcn_wmma_f32_16x16x32_f16(
          /*neg_a=*/false, afrag, /*neg_b=*/false, bfrag,
          /*c_mod=*/(short)0, acc, /*reuse_a=*/false, /*reuse_b=*/false);
    }
  }

  // Epilogue: C/D layout — VGPR r holds row (r + 8*khigh), column = lane&15.
  const float bv = bias[oc];
#pragma unroll
  for (int r = 0; r < 8; ++r) {
    const int m = r + 8 * khigh;
    const int p = tile_m * 16 + m;
    if (p < M) {
      const int pb = p / hw;                     // compile-time divisors
      const int pr = p % hw;
      const int y  = pr / Wout;
      const int x  = pr % Wout;
      float v = acc[r] + bv;
      v = v > 0.f ? v : 0.f;                     // ReLU
      outNCHW[(((long)pb * Cout + oc) * Hout + y) * Wout + x] = v;
      if (outNHWC)                               // padded NHWC f16 for next conv
        outNHWC[(((long)pb * (Hout + 2) + y + 1) * (Wout + 2) + (x + 1)) * Cout + oc] =
            (_Float16)v;
    }
  }
}

// ---------------------------------------------------------------------------
// 1x1 score heads: one thread per (batch, anchor) -> 128-wide dot product.
// ---------------------------------------------------------------------------
__global__ void rpn_scores_kernel(const float* __restrict__ d1,
                                  const float* __restrict__ d2,
                                  const float* __restrict__ d3,
                                  const float* __restrict__ wt1, const float* __restrict__ bt1,
                                  const float* __restrict__ wt2, const float* __restrict__ bt2,
                                  const float* __restrict__ wt3, const float* __restrict__ bt3,
                                  float* __restrict__ scores) {
  const int t = blockIdx.x * blockDim.x + threadIdx.x;
  if (t >= BATCH * A_N) return;
  const int b = t / A_N;
  const int a = t % A_N;

  const float* feat; const float* w; float bv; int hw;
  if (a < 1176) {
    const int c = a / 196, pos = a % 196;
    feat = d1 + (long)b * 128 * 196 + pos; w = wt1 + c * 128; bv = bt1[c]; hw = 196;
  } else if (a < 1470) {
    const int aa = a - 1176, c = aa / 49, pos = aa % 49;
    feat = d2 + (long)b * 128 * 49 + pos; w = wt2 + c * 128; bv = bt2[c]; hw = 49;
  } else {
    const int aa = a - 1470, c = aa / 16, pos = aa % 16;
    feat = d3 + (long)b * 128 * 16 + pos; w = wt3 + c * 128; bv = bt3[c]; hw = 16;
  }
  float s = bv;
#pragma unroll 4
  for (int ic = 0; ic < 128; ++ic) s += feat[(long)ic * hw] * w[ic];
  scores[t] = s;
}

// ---------------------------------------------------------------------------
// Greedy NMS + top-6 per batch. One 256-thread workgroup per batch image,
// everything staged in LDS (~53 KB of the 320 KB WGP pool).
// ---------------------------------------------------------------------------
__global__ void nms_topk_kernel(const float* __restrict__ scores,
                                const float* __restrict__ anchors,  // (A,4) y0,x0,y1,x1
                                float* __restrict__ dout) {
  __shared__ float ss[A_N];
  __shared__ float sy0[A_N], sx0[A_N], sy1[A_N], sx1[A_N], sarea[A_N];
  __shared__ float masked[A_N];
  __shared__ int   order[A_N];
  __shared__ unsigned char supp[A_N];

  const int b   = blockIdx.x;
  const int tid = threadIdx.x;
  const int nt  = blockDim.x;

  for (int i = tid; i < A_N; i += nt) {
    ss[i]   = scores[b * A_N + i];
    supp[i] = 0;
  }
  __syncthreads();

  // Stable descending rank (matches jnp.argsort(-s): ties -> lower index first)
  for (int i = tid; i < A_N; i += nt) {
    const float si = ss[i];
    int rk = 0;
    for (int j = 0; j < A_N; ++j) {
      const float sj = ss[j];
      rk += (sj > si) || (sj == si && j < i);
    }
    order[rk] = i;
  }
  __syncthreads();

  for (int r = tid; r < A_N; r += nt) {
    const int a = order[r];
    const float y0 = anchors[a * 4 + 0], x0 = anchors[a * 4 + 1];
    const float y1 = anchors[a * 4 + 2], x1 = anchors[a * 4 + 3];
    sy0[r] = y0; sx0[r] = x0; sy1[r] = y1; sx1[r] = x1;
    sarea[r] = (y1 - y0) * (x1 - x0);
  }
  __syncthreads();

  // Serial-dependent greedy suppression; inner row parallel over the block.
  for (int i = 0; i < A_N - 1; ++i) {
    __syncthreads();
    if (supp[i]) continue;                 // uniform across block
    const float y0 = sy0[i], x0 = sx0[i], y1 = sy1[i], x1 = sx1[i], ar = sarea[i];
    for (int j = i + 1 + tid; j < A_N; j += nt) {
      float ih = fminf(y1, sy1[j]) - fmaxf(y0, sy0[j]); ih = ih > 0.f ? ih : 0.f;
      float iw = fminf(x1, sx1[j]) - fmaxf(x0, sx0[j]); iw = iw > 0.f ? iw : 0.f;
      const float inter = ih * iw;
      const float iou   = inter / (ar + sarea[j] - inter);
      if (iou > 0.25f) supp[j] = 1;
    }
  }
  __syncthreads();

  for (int r = tid; r < A_N; r += nt)
    masked[order[r]] = supp[r] ? -__builtin_inff() : ss[order[r]];
  __syncthreads();

  if (tid == 0) {
    float* coords = dout + (long)BATCH * TOPN * 3 * OUT_PX * OUT_PX;   // 7225344
    float* prob   = coords + BATCH * TOPN * 4;
    float* idxo   = prob + BATCH * TOPN;
    for (int t = 0; t < TOPN; ++t) {
      float best = -__builtin_inff(); int bi = 0;
      for (int a = 0; a < A_N; ++a)
        if (masked[a] > best) { best = masked[a]; bi = a; }
      masked[bi] = -__builtin_inff();
      prob[b * TOPN + t] = best;
      idxo[b * TOPN + t] = (float)bi;
      const float y0 = anchors[bi * 4 + 0], x0 = anchors[bi * 4 + 1];
      const float y1 = anchors[bi * 4 + 2], x1 = anchors[bi * 4 + 3];
      float* c = coords + (long)(b * TOPN + t) * 4;
      c[0] = x0; c[1] = y0; c[2] = x1; c[3] = y1;   // (x0,y0,x1,y1)
    }
  }
}

// ---------------------------------------------------------------------------
// Bilinear crop-resize in padded coordinate space; pad never materialized.
// ---------------------------------------------------------------------------
__global__ void crop_resize_kernel(const float* __restrict__ x,
                                   const float* __restrict__ coords,  // (B*TOPN,4) x0,y0,x1,y1
                                   float* __restrict__ out) {
  const int blk = blockIdx.x;          // bn*3 + c
  const int c   = blk % 3;
  const int bn  = blk / 3;             // b*TOPN + n
  const int b   = bn / TOPN;

  const float* co = coords + (long)bn * 4;
  const int x0 = (int)co[0], y0 = (int)co[1], x1 = (int)co[2], y1 = (int)co[3];

  const float* img = x + ((long)b * 3 + c) * IMG_PX * IMG_PX;
  float* o = out + ((long)bn * 3 + c) * OUT_PX * OUT_PX;

  const float dy = (float)(y1 - 1 - y0);
  const float dx = (float)(x1 - 1 - x0);

  for (int p = threadIdx.x; p < OUT_PX * OUT_PX; p += blockDim.x) {
    const int oy = p / OUT_PX, ox = p % OUT_PX;
    const float sy = (float)y0 + ((float)oy / (float)(OUT_PX - 1)) * dy;
    const float sx = (float)x0 + ((float)ox / (float)(OUT_PX - 1)) * dx;
    const int y0i = (int)floorf(sy), x0i = (int)floorf(sx);
    const int y1i = min(y0i + 1, y1 - 1), x1i = min(x0i + 1, x1 - 1);
    const float wy = sy - (float)y0i, wx = sx - (float)x0i;

    auto g = [&](int yi, int xi) -> float {
      const int yy = yi - PAD_PX, xx = xi - PAD_PX;
      if (yy < 0 || yy >= IMG_PX || xx < 0 || xx >= IMG_PX) return 0.f;
      return img[(long)yy * IMG_PX + xx];
    };
    const float top = (1.f - wx) * g(y0i, x0i) + wx * g(y0i, x1i);
    const float bot = (1.f - wx) * g(y1i, x0i) + wx * g(y1i, x1i);
    o[p] = (1.f - wy) * top + wy * bot;
  }
}

// ---------------------------------------------------------------------------
extern "C" void kernel_launch(void* const* d_in, const int* in_sizes, int n_in,
                              void* d_out, int out_size, void* d_ws, size_t ws_size,
                              hipStream_t stream) {
  const float* x       = (const float*)d_in[0];
  const float* rpn     = (const float*)d_in[1];
  const float* anchors = (const float*)d_in[2];
  const float* wd1 = (const float*)d_in[3];  const float* bd1 = (const float*)d_in[4];
  const float* wd2 = (const float*)d_in[5];  const float* bd2 = (const float*)d_in[6];
  const float* wd3 = (const float*)d_in[7];  const float* bd3 = (const float*)d_in[8];
  const float* wt1 = (const float*)d_in[9];  const float* bt1 = (const float*)d_in[10];
  const float* wt2 = (const float*)d_in[11]; const float* bt2 = (const float*)d_in[12];
  const float* wt3 = (const float*)d_in[13]; const float* bt3 = (const float*)d_in[14];
  (void)in_sizes; (void)n_in; (void)out_size; (void)ws_size;

  // ---- workspace layout --------------------------------------------------
  // f16 region (zero-padded activation buffers first, so one zero pass covers them)
  _Float16* aP1 = (_Float16*)d_ws;                   // (8,16,16,2048) = 4,194,304 h
  _Float16* d1P = aP1 + (long)8 * 16 * 16 * 2048;    // (8,16,16,128)  =   262,144 h
  _Float16* d2P = d1P + (long)8 * 16 * 16 * 128;     // (8, 9, 9,128)  =    82,944 h
  const long nZero = (long)8 * 16 * 16 * 2048 + (long)8 * 16 * 16 * 128 + (long)8 * 9 * 9 * 128;
  _Float16* wP1 = d2P + (long)8 * 9 * 9 * 128;       // 9*128*2048 = 2,359,296 h
  _Float16* wP2 = wP1 + (long)9 * 128 * 2048;        // 9*128*128  =   147,456 h
  _Float16* wP3 = wP2 + (long)9 * 128 * 128;         //                147,456 h
  // f32 region
  float* d1 = (float*)(wP3 + (long)9 * 128 * 128);   // 8*128*196 = 200,704 f
  float* d2 = d1 + 200704;                           // 8*128*49  =  50,176 f
  float* d3 = d2 + 50176;                            // 8*128*16  =  16,384 f
  float* sc = d3 + 16384;                            // 8*1614    =  12,912 f
  float* outf = (float*)d_out;

  // ---- pack --------------------------------------------------------------
  zero_f16<<<(int)((nZero + 255) / 256), 256, 0, stream>>>(aP1, nZero);
  pack_act<2048, 14, 14><<<(8 * 2048 * 196 + 255) / 256, 256, 0, stream>>>(rpn, aP1);
  pack_wgt<2048, 128><<<(128 * 2048 * 9 + 255) / 256, 256, 0, stream>>>(wd1, wP1);
  pack_wgt<128, 128><<<(128 * 128 * 9 + 255) / 256, 256, 0, stream>>>(wd2, wP2);
  pack_wgt<128, 128><<<(128 * 128 * 9 + 255) / 256, 256, 0, stream>>>(wd3, wP3);

  // ---- convs (WMMA implicit GEMM) ---------------------------------------
  // conv1: M=1568 -> 98 M-tiles x 8 N-tiles, K = 2048 per tap
  conv3x3_wmma<2048, 16, 16, 128, 14, 14, 1><<<98 * 8, 32, 0, stream>>>(
      aP1, wP1, bd1, d1, d1P);
  // conv2: M=392 -> 25 M-tiles (stride 2)
  conv3x3_wmma<128, 16, 16, 128, 7, 7, 2><<<25 * 8, 32, 0, stream>>>(
      d1P, wP2, bd2, d2, d2P);
  // conv3: M=128 -> 8 M-tiles (stride 2)
  conv3x3_wmma<128, 9, 9, 128, 4, 4, 2><<<8 * 8, 32, 0, stream>>>(
      d2P, wP3, bd3, d3, (_Float16*)nullptr);

  // ---- heads / NMS / crop ------------------------------------------------
  rpn_scores_kernel<<<(BATCH * A_N + 255) / 256, 256, 0, stream>>>(
      d1, d2, d3, wt1, bt1, wt2, bt2, wt3, bt3, sc);
  nms_topk_kernel<<<BATCH, 256, 0, stream>>>(sc, anchors, outf);
  const float* coords = outf + (long)BATCH * TOPN * 3 * OUT_PX * OUT_PX;
  crop_resize_kernel<<<BATCH * TOPN * 3, 256, 0, stream>>>(x, coords, outf);
}
